// AKT_23613730193806
// MI455X (gfx1250) — matrix operand order repaired and checked by reading, not compiled, not run on hardware
//
#include <hip/hip_runtime.h>

// =====================================================================
// AKT forward for gfx1250 (MI455X).
//  - All GEMMs: V_WMMA_F32_16X16X32_F16, f16 operands, f32 accumulate.
//  - Weights pre-converted ONCE per launch to transposed f16 Wt[N][K]
//    (halves L2 traffic, kills inner-loop cvt, makes tiles rectangular).
//  - GEMM tiles staged LDS via Tensor Data Mover (tensor_load_to_lds,
//    D# per ISA 8.3/8.4, LDS padding 16B/64B), double-buffered with
//    s_wait_tensorcnt.  Toolchain uses the 6-arg builtin form:
//    (u32x4 g0, i32x8 g1, i32x4 g2, i32x4 g3, i32x8 extra, i32 cpol).
//  - Attention: WMMA QK^T straight from global f16 (b128 fragments),
//    wave32 shuffle softmax/cumsum/decay, WMMA P@V from LDS.
//
// Input order (setup_inputs() dict insertion order):
//   0:Ec 1:Ecv 2:Eiv 3:Ei 4:Eqd
//   5..19 enc1[0] (Wqk,bqk,Wv,bv,Wo,bo,gamma,ln1s,ln1b,W1,b1,W2,b2,ln2s,ln2b)
//   20..34 enc1[1]; 35..49 enc2[0]; 50..64 enc2[1]; 65..79 enc2[2]; 80..94 enc2[3]
//   95..100 head (W1,b1,W2,b2,W3,b3); 101:concept 102:question 103:correct
// =====================================================================

typedef __attribute__((ext_vector_type(16))) _Float16 v16h;
typedef __attribute__((ext_vector_type(8)))  float    v8f;
typedef __attribute__((ext_vector_type(4)))  unsigned u32x4;
typedef __attribute__((ext_vector_type(8)))  int      i32x8;
typedef __attribute__((ext_vector_type(4)))  int      i32x4;

union U16 { v16h v; unsigned u[8]; };

#define NTOK 8192
#define DMODEL 512
#define NHEAD 8
#define DHEAD 64
#define SEQL 512
#define BATCH 16
#define FFDIM 2048
#define NCONC 1024

#define LDSPITCH 40          // halfs per LDS tile row (64B data + 16B pad)

#if __has_builtin(__builtin_amdgcn_tensor_load_to_lds)
#define HAVE_TDM 1
#else
#define HAVE_TDM 0
#endif

#if __has_builtin(__builtin_amdgcn_s_wait_tensorcnt)
#define WAIT_TENSOR(n) __builtin_amdgcn_s_wait_tensorcnt((short)(n))
#else
#define WAIT_TENSOR(n) asm volatile("s_wait_tensorcnt %0" :: "n"(n) : "memory")
#endif

// ---------------- wave32 helpers ----------------
__device__ __forceinline__ float wredsum(float v){
#pragma unroll
  for (int m = 16; m > 0; m >>= 1) v += __shfl_xor(v, m, 32);
  return v;
}
__device__ __forceinline__ float wredmax(float v){
#pragma unroll
  for (int m = 16; m > 0; m >>= 1) v = fmaxf(v, __shfl_xor(v, m, 32));
  return v;
}
__device__ __forceinline__ float wexscan(float v, int lane){
  float t = v;
#pragma unroll
  for (int off = 1; off < 32; off <<= 1){
    float n = __shfl_up(t, off, 32);
    if (lane >= off) t += n;
  }
  return t - v;
}
__device__ __forceinline__ v8f wmma16(const v16h a, const v16h b, const v8f c){
  return __builtin_amdgcn_wmma_f32_16x16x32_f16(false, a, false, b, (short)0, c, false, false);
}

#if HAVE_TDM
// ---- Tensor DMA descriptor (ISA 8.3/8.4): 2-D f16 tile, LDS padding ----
struct TdmDesc { u32x4 g0; i32x8 g1; i32x4 g2; i32x4 g3; };
__device__ __forceinline__ TdmDesc tdm_make(const void* gaddr, unsigned ldsOff,
                                            unsigned tileRows, unsigned strideElems){
  unsigned long long ga = (unsigned long long)gaddr;
  TdmDesc d;
  d.g0[0] = 1u;                                              // count=1, user D#
  d.g0[1] = ldsOff;                                          // lds_addr (bytes)
  d.g0[2] = (unsigned)ga;                                    // global_addr[31:0]
  d.g0[3] = (unsigned)((ga >> 32) & 0x01FFFFFFu) | (2u<<30); // ga[56:32] | type=2
  unsigned td0 = strideElems;         // tensor_dim0 (elements along K)
  unsigned td1 = 0x100000u;           // tensor_dim1: large (tile fully in-bounds)
  unsigned long long s0 = strideElems;
  d.g1[0] = (int)(0x10000u            // data_size = 2 bytes
                | (1u << 20)          // pad_enable
                | (3u << 22)          // pad_interval: every 16 DWORDs (64B)
                | (3u << 25));        // pad_amount: 4 DWORDs (16B) -> 80B pitch
  d.g1[1] = (int)((td0 & 0xFFFFu) << 16);
  d.g1[2] = (int)((td0 >> 16) | ((td1 & 0xFFFFu) << 16));
  d.g1[3] = (int)((td1 >> 16) | (32u << 16));                // tile_dim0 = 32
  d.g1[4] = (int)(tileRows);                                 // tile_dim1; tile_dim2=0
  d.g1[5] = (int)(unsigned)(s0 & 0xFFFFFFFFu);               // dim0_stride lo
  d.g1[6] = (int)(unsigned)(s0 >> 32);                       // dim0_stride hi
  d.g1[7] = 0;
  i32x4 z = {0,0,0,0};
  d.g2 = z; d.g3 = z;                                        // 2-D: groups 2/3 unused
  return d;
}
__device__ __forceinline__ void tdm_issue(const TdmDesc& d){
  i32x8 z8 = {0,0,0,0,0,0,0,0};
  __builtin_amdgcn_tensor_load_to_lds(d.g0, d.g1, d.g2, d.g3, z8, 0);
}
#endif

// ---------------- Rasch embedding ----------------
__global__ __launch_bounds__(256)
void embed_kernel(const float* __restrict__ Ec,  const float* __restrict__ Ecv,
                  const float* __restrict__ Eiv, const float* __restrict__ Ei,
                  const float* __restrict__ Eqd,
                  const int* __restrict__ cseq, const int* __restrict__ qseq,
                  const int* __restrict__ rseq,
                  float* __restrict__ Y, _Float16* __restrict__ Yh,
                  float* __restrict__ QE, _Float16* __restrict__ QEh)
{
  int t = blockIdx.x;
  int c = cseq[t], q = qseq[t], r = rseq[t];
  float qd = Eqd[q];
  for (int col = threadIdx.x; col < DMODEL; col += blockDim.x){
    float ce = Ec [(size_t)c * DMODEL + col];
    float cv = Ecv[(size_t)c * DMODEL + col];
    float iv = Eiv[((size_t)c + (size_t)NCONC * r) * DMODEL + col];
    float ei = Ei [(size_t)r * DMODEL + col];
    float qe = ce + qd * cv;
    float ie = ei + ce + qd * (iv + cv);
    QE [(size_t)t * DMODEL + col] = qe;
    QEh[(size_t)t * DMODEL + col] = (_Float16)qe;
    Y  [(size_t)t * DMODEL + col] = ie;
    Yh [(size_t)t * DMODEL + col] = (_Float16)ie;
  }
}

// ---------------- weight convert+transpose: W[K][N] f32 -> Wt[N][K] f16 --
__global__ __launch_bounds__(256)
void conv_wt_kernel(const float* __restrict__ W, _Float16* __restrict__ Wt,
                    int K, int N)
{
  int i = blockIdx.x * 256 + threadIdx.x;          // over K*N, coalesced read
  int k = i / N, n = i % N;
  Wt[(size_t)n * K + k] = (_Float16)W[i];
}

// ---------------- WMMA GEMM:  Out = act(A[M,K]f16 @ Wt[N,K]^T + bias) ----
// 256 threads = 8 waves; block tile 128x64; wave tile 32x32; K-step 32.
__device__ __forceinline__ void gemm_step(const _Float16* __restrict__ As_,
                                          const _Float16* __restrict__ Bs_,
                                          int wm, int wn, int ksel, int mrow,
                                          v8f (&acc)[2][2])
{
  U16 a[2], b[2];
#pragma unroll
  for (int mi = 0; mi < 2; ++mi){
    const _Float16* p = As_ + (size_t)(wm + mi*16 + mrow) * LDSPITCH;
#pragma unroll
    for (int v = 0; v < 8; ++v)
      a[mi].u[v] = *(const unsigned*)(p + ksel*16 + 2*v);
  }
#pragma unroll
  for (int ni = 0; ni < 2; ++ni){
    const _Float16* p = Bs_ + (size_t)(wn + ni*16 + mrow) * LDSPITCH;
#pragma unroll
    for (int v = 0; v < 8; ++v)
      b[ni].u[v] = *(const unsigned*)(p + ksel*16 + 2*v);
  }
#pragma unroll
  for (int mi = 0; mi < 2; ++mi)
#pragma unroll
    for (int ni = 0; ni < 2; ++ni)
      acc[mi][ni] = wmma16(a[mi].v, b[ni].v, acc[mi][ni]);
}

template<int OUT16, int ACT>
__global__ __launch_bounds__(256)
void gemm_wmma_k(const _Float16* __restrict__ A, const _Float16* __restrict__ Wt,
                 const float* __restrict__ bias, void* __restrict__ Out,
                 int M, int N, int K)
{
  __shared__ __align__(16) _Float16 As[2][128 * LDSPITCH];   // 80B row pitch
  __shared__ __align__(16) _Float16 Bs[2][64  * LDSPITCH];
  const int bm = blockIdx.y * 128, bn = blockIdx.x * 64;
  const int tid = threadIdx.x, lane = tid & 31, wave = tid >> 5;
  const int wm = (wave & 3) * 32, wn = (wave >> 2) * 32;
  const int ksel = (lane >> 4) & 1, mrow = lane & 15;
  v8f zv = {0.f,0.f,0.f,0.f,0.f,0.f,0.f,0.f};
  v8f acc[2][2];
#pragma unroll
  for (int i = 0; i < 2; ++i)
#pragma unroll
    for (int j = 0; j < 2; ++j) acc[i][j] = zv;

  const int nsteps = K >> 5;

#if HAVE_TDM
  // --- TDM double-buffered pipeline: DMA tile t+1 while WMMA on tile t ---
  const unsigned asOff =
      (unsigned)(unsigned long long)(__attribute__((address_space(3))) _Float16*)&As[0][0];
  const unsigned bsOff =
      (unsigned)(unsigned long long)(__attribute__((address_space(3))) _Float16*)&Bs[0][0];
  const unsigned asBytes = 128u * LDSPITCH * 2u;
  const unsigned bsBytes = 64u  * LDSPITCH * 2u;

  if (wave == 0){
    tdm_issue(tdm_make(A  + (size_t)bm * K, asOff, 128, (unsigned)K));
    tdm_issue(tdm_make(Wt + (size_t)bn * K, bsOff, 64,  (unsigned)K));
  }
  for (int kt = 0; kt < nsteps; ++kt){
    if (wave == 0){
      if (kt + 1 < nsteps){
        int nb = (kt + 1) & 1, k0 = (kt + 1) << 5;
        tdm_issue(tdm_make(A  + (size_t)bm * K + k0, asOff + nb*asBytes, 128, (unsigned)K));
        tdm_issue(tdm_make(Wt + (size_t)bn * K + k0, bsOff + nb*bsBytes, 64, (unsigned)K));
        WAIT_TENSOR(2);          // in-order per-wave: older pair complete
      } else {
        WAIT_TENSOR(0);
      }
    }
    __syncthreads();
    gemm_step(&As[kt & 1][0], &Bs[kt & 1][0], wm, wn, ksel, mrow, acc);
    __syncthreads();             // before DMA overwrites this buffer
  }
#else
  // --- synchronous fallback staging ---
  for (int kt = 0; kt < nsteps; ++kt){
    int k0 = kt << 5;
    __syncthreads();
    for (int i = tid; i < 128*32; i += 256){
      int m = i >> 5, k = i & 31;
      As[0][m * LDSPITCH + k] = A[(size_t)(bm + m) * K + k0 + k];
    }
    for (int i = tid; i < 64*32; i += 256){
      int n = i >> 5, k = i & 31;
      Bs[0][n * LDSPITCH + k] = Wt[(size_t)(bn + n) * K + k0 + k];
    }
    __syncthreads();
    gemm_step(&As[0][0], &Bs[0][0], wm, wn, ksel, mrow, acc);
  }
#endif

#pragma unroll
  for (int mi = 0; mi < 2; ++mi)
#pragma unroll
    for (int ni = 0; ni < 2; ++ni){
      int col = bn + wn + ni*16 + mrow;
      float bv = bias ? bias[col] : 0.f;
#pragma unroll
      for (int r = 0; r < 8; ++r){
        int row = bm + wm + mi*16 + ksel*8 + r;
        float val = acc[mi][ni][r] + bv;
        if (ACT) val = fmaxf(val, 0.f);
        if (OUT16) ((_Float16*)Out)[(size_t)row * N + col] = (_Float16)val;
        else       ((float*)Out)   [(size_t)row * N + col] = val;
      }
    }
}

// ---------------- AKT monotonic attention (q==k, shared projection) ------
// One workgroup per (b, h, 64-query tile).
// Dynamic LDS: S f32[64][512] | P f16[64][512] | Vt f16[64][512] = 256KB.
#define ATTN_LDS (64*512*4 + 64*512*2 + 64*512*2)
__global__ __launch_bounds__(256)
void attn_kernel(const _Float16* __restrict__ QK, const _Float16* __restrict__ V,
                 _Float16* __restrict__ O, const float* __restrict__ gamma,
                 int excl, int zp)
{
  extern __shared__ char smem[];
  float*    S  = (float*)smem;
  _Float16* P  = (_Float16*)(smem + 64*512*4);
  _Float16* Vt = (_Float16*)(smem + 64*512*4 + 64*512*2);
  const int qb = blockIdx.x * 64;
  const int hh = blockIdx.y, bb = blockIdx.z;
  const int tid = threadIdx.x, lane = tid & 31, wave = tid >> 5;
  const int ksel = (lane >> 4) & 1, mrow = lane & 15;

  float gm = gamma[hh];
  float gdec = -(fmaxf(gm, 0.f) + log1pf(expf(-fabsf(gm))));

  const _Float16* Qb = QK + (size_t)bb * SEQL * DMODEL + hh * DHEAD;

  // Phase 1: S = Q @ K^T / 8 ; wave w owns 64-key block, fragments from global
  {
    const int cb = wave * 64;
    v8f zv = {0.f,0.f,0.f,0.f,0.f,0.f,0.f,0.f};
    v8f acc[4][4];
#pragma unroll
    for (int i = 0; i < 4; ++i)
#pragma unroll
      for (int j = 0; j < 4; ++j) acc[i][j] = zv;

    for (int kk = 0; kk < DHEAD; kk += 32){
      U16 a[4], bfr[4];
#pragma unroll
      for (int mi = 0; mi < 4; ++mi){
        const _Float16* p = Qb + (size_t)(qb + mi*16 + mrow) * DMODEL + kk;
#pragma unroll
        for (int v = 0; v < 8; ++v)
          a[mi].u[v] = *(const unsigned*)(p + ksel*16 + 2*v);
      }
#pragma unroll
      for (int ni = 0; ni < 4; ++ni){
        const _Float16* p = Qb + (size_t)(cb + ni*16 + mrow) * DMODEL + kk;
#pragma unroll
        for (int v = 0; v < 8; ++v)
          bfr[ni].u[v] = *(const unsigned*)(p + ksel*16 + 2*v);
      }
#pragma unroll
      for (int mi = 0; mi < 4; ++mi)
#pragma unroll
        for (int ni = 0; ni < 4; ++ni)
          acc[mi][ni] = wmma16(a[mi].v, bfr[ni].v, acc[mi][ni]);
    }
#pragma unroll
    for (int mi = 0; mi < 4; ++mi)
#pragma unroll
      for (int ni = 0; ni < 4; ++ni)
#pragma unroll
        for (int r = 0; r < 8; ++r)
          S[(mi*16 + ksel*8 + r) * 512 + cb + ni*16 + mrow] = acc[mi][ni][r] * 0.125f;
  }
  __syncthreads();

  // stage V^T into LDS
  {
    const _Float16* Vb = V + (size_t)bb * SEQL * DMODEL + hh * DHEAD;
    for (int i = tid; i < 64*512; i += 256){
      int dh = i & 63, j = i >> 6;
      Vt[dh * 512 + j] = Vb[(size_t)j * DMODEL + dh];
    }
  }

  // Phase 2: masked softmax -> cumsum -> distance decay -> softmax -> P f16
  for (int rr = 0; rr < 8; ++rr){
    int r = wave * 8 + rr;
    int qi = qb + r;
    int cb2 = lane * 16;
    float sreg[16], e[16];
    float mx = -1e30f;
#pragma unroll
    for (int jj = 0; jj < 16; ++jj){
      int j = cb2 + jj;
      float s = S[r * 512 + j];
      sreg[jj] = s;
      bool ok = excl ? (j < qi) : (j <= qi);
      if (ok && s > mx) mx = s;
    }
    mx = wredmax(mx);
    float lsum = 0.f;
#pragma unroll
    for (int jj = 0; jj < 16; ++jj){
      int j = cb2 + jj;
      bool ok = excl ? (j < qi) : (j <= qi);
      float ev = ok ? expf(sreg[jj] - mx) : 0.f;
      e[jj] = ev; lsum += ev;
    }
    float tot = wredsum(lsum);
    float inv = tot > 0.f ? 1.f / tot : 0.f;
    float pre = wexscan(lsum * inv, lane);
    float disttot = tot > 0.f ? 1.f : 0.f;
    float run = pre, mx2 = -1e30f;
#pragma unroll
    for (int jj = 0; jj < 16; ++jj){
      int j = cb2 + jj;
      bool ok = excl ? (j < qi) : (j <= qi);
      run += e[jj] * inv;
      float pos = fabsf((float)(qi - j));
      float dd = sqrtf(fmaxf((disttot - run) * pos, 0.f));
      float eff = fminf(fmaxf(expf(dd * gdec), 1e-5f), 1e5f);
      float s2 = ok ? sreg[jj] * eff : -1e30f;
      sreg[jj] = s2;
      if (ok && s2 > mx2) mx2 = s2;
    }
    mx2 = wredmax(mx2);
    float lsum2 = 0.f;
#pragma unroll
    for (int jj = 0; jj < 16; ++jj){
      int j = cb2 + jj;
      bool ok = excl ? (j < qi) : (j <= qi);
      float ev = ok ? expf(sreg[jj] - mx2) : 0.f;
      sreg[jj] = ev; lsum2 += ev;
    }
    float tot2 = wredsum(lsum2);
    float inv2 = tot2 > 0.f ? 1.f / tot2 : 0.f;
#pragma unroll
    for (int jj = 0; jj < 16; ++jj){
      float pv = sreg[jj] * inv2;
      if (zp && qi == 0) pv = 0.f;
      P[r * 512 + cb2 + jj] = (_Float16)pv;
    }
  }
  __syncthreads();

  // Phase 3: out = P @ V (f16 out feeds the Wo GEMM directly)
  {
    const int rt = wave & 3, ch = wave >> 2;
    v8f zv = {0.f,0.f,0.f,0.f,0.f,0.f,0.f,0.f};
    v8f o0 = zv, o1 = zv;
    for (int j0 = 0; j0 < 512; j0 += 32){
      U16 a, b0, b1;
      const _Float16* pr = P + (size_t)(rt*16 + mrow) * 512 + j0;
#pragma unroll
      for (int v = 0; v < 8; ++v)
        a.u[v] = *(const unsigned*)(pr + ksel*16 + 2*v);
      const _Float16* p0 = Vt + (size_t)(ch*32 +  0 + mrow) * 512 + j0;
      const _Float16* p1 = Vt + (size_t)(ch*32 + 16 + mrow) * 512 + j0;
#pragma unroll
      for (int v = 0; v < 8; ++v){
        b0.u[v] = *(const unsigned*)(p0 + ksel*16 + 2*v);
        b1.u[v] = *(const unsigned*)(p1 + ksel*16 + 2*v);
      }
      o0 = wmma16(a.v, b0.v, o0);
      o1 = wmma16(a.v, b1.v, o1);
    }
    _Float16* Ob = O + (size_t)bb * SEQL * DMODEL + hh * DHEAD;
#pragma unroll
    for (int r = 0; r < 8; ++r){
      int qi = qb + rt*16 + ksel*8 + r;
      Ob[(size_t)qi * DMODEL + ch*32 +  0 + mrow] = (_Float16)o0[r];
      Ob[(size_t)qi * DMODEL + ch*32 + 16 + mrow] = (_Float16)o1[r];
    }
  }
}

// ---------------- residual + layernorm; writes f32 stream + f16 mirror ---
__global__ __launch_bounds__(256)
void ln_kernel(float* __restrict__ x, _Float16* __restrict__ xh,
               const float* __restrict__ f,
               const float* __restrict__ s, const float* __restrict__ b)
{
  int lane = threadIdx.x & 31, wave = threadIdx.x >> 5;
  size_t t = (size_t)blockIdx.x * 8 + wave;
  float v[16];
  float sum = 0.f;
#pragma unroll
  for (int k = 0; k < 16; ++k){
    int c = lane + 32*k;
    float z = x[t * DMODEL + c] + f[t * DMODEL + c];
    v[k] = z; sum += z;
  }
  sum = wredsum(sum);
  float mean = sum * (1.f / DMODEL);
  float var = 0.f;
#pragma unroll
  for (int k = 0; k < 16; ++k){ float d = v[k] - mean; var += d * d; }
  var = wredsum(var) * (1.f / DMODEL);
  float rs = rsqrtf(var + 1e-5f);
#pragma unroll
  for (int k = 0; k < 16; ++k){
    int c = lane + 32*k;
    float o = (v[k] - mean) * rs * s[c] + b[c];
    x [t * DMODEL + c] = o;
    xh[t * DMODEL + c] = (_Float16)o;
  }
}

// ---------------- concat [x | q_emb] -> f16 ----------------
__global__ __launch_bounds__(256)
void concat_kernel(const float* __restrict__ X, const float* __restrict__ QE,
                   _Float16* __restrict__ H)
{
  int i = blockIdx.x * blockDim.x + threadIdx.x;   // over NTOK*512
  int t = i >> 9, c = i & 511;
  H[(size_t)t * 1024 + c]       = (_Float16)X[i];
  H[(size_t)t * 1024 + 512 + c] = (_Float16)QE[i];
}

// ---------------- final 256-dot + sigmoid ----------------
__global__ __launch_bounds__(256)
void head_final_kernel(const _Float16* __restrict__ Hin, const float* __restrict__ W3,
                       const float* __restrict__ b3, float* __restrict__ out)
{
  int lane = threadIdx.x & 31, wave = threadIdx.x >> 5;
  size_t t = (size_t)blockIdx.x * 8 + wave;
  float acc = 0.f;
#pragma unroll
  for (int k = 0; k < 8; ++k){
    int c = lane + 32*k;
    acc += (float)Hin[t * 256 + c] * W3[c];
  }
  acc = wredsum(acc);
  if (lane == 0) out[t] = 1.f / (1.f + expf(-(acc + b3[0])));
}

// =====================================================================
extern "C" void kernel_launch(void* const* d_in, const int* in_sizes, int n_in,
                              void* d_out, int out_size, void* d_ws, size_t ws_size,
                              hipStream_t stream)
{
  (void)in_sizes; (void)n_in; (void)out_size; (void)ws_size;
  auto fp = [&](int i){ return (const float*)d_in[i]; };

  struct LayerP {
    const float *Wqk,*bqk,*Wv,*bv,*Wo,*bo,*gamma,*ln1s,*ln1b,*W1,*b1,*W2,*b2,*ln2s,*ln2b;
  };
  auto layer = [&](int base){
    LayerP p = { fp(base+0), fp(base+1), fp(base+2), fp(base+3), fp(base+4),
                 fp(base+5), fp(base+6), fp(base+7), fp(base+8), fp(base+9),
                 fp(base+10), fp(base+11), fp(base+12), fp(base+13), fp(base+14) };
    return p;
  };
  LayerP enc1[2] = { layer(5),  layer(20) };
  LayerP enc2[4] = { layer(35), layer(50), layer(65), layer(80) };
  const float* hW1 = fp(95); const float* hb1 = fp(96);
  const float* hW2 = fp(97); const float* hb2 = fp(98);
  const float* hW3 = fp(99); const float* hb3 = fp(100);
  const int* cseq = (const int*)d_in[101];
  const int* qseq = (const int*)d_in[102];
  const int* rseq = (const int*)d_in[103];

  // ---- workspace carve ----
  const size_t SZF = (size_t)NTOK * DMODEL * sizeof(float);      // 16 MB
  const size_t SZH = (size_t)NTOK * DMODEL * sizeof(_Float16);   //  8 MB
  char* w = (char*)d_ws; size_t off = 0;
  auto alloc = [&](size_t bytes){ void* p = w + off; off += (bytes + 255) & ~(size_t)255; return p; };
  float*    Y    = (float*)alloc(SZF);
  float*    X    = (float*)alloc(SZF);
  float*    QE   = (float*)alloc(SZF);
  float*    A1   = (float*)alloc(SZF);                     // f32 GEMM outs (residual adds)
  _Float16* Yh   = (_Float16*)alloc(SZH);
  _Float16* Xh   = (_Float16*)alloc(SZH);
  _Float16* QEh  = (_Float16*)alloc(SZH);
  _Float16* QH   = (_Float16*)alloc(SZH);                  // q==k projection
  _Float16* VH   = (_Float16*)alloc(SZH);
  _Float16* AOh  = (_Float16*)alloc(SZH);                  // attention out f16
  _Float16* F16a = (_Float16*)alloc((size_t)NTOK * FFDIM * 2);   // hidden / hcat
  _Float16* WTar = (_Float16*)alloc((size_t)14000000 * 2);       // f16 weight arena

  // ---- pre-convert weights once: W[K][N] f32 -> Wt[N][K] f16 ----
  _Float16* wt_next = WTar;
  auto prep = [&](const float* Ws, int K, int N){
    _Float16* dst = wt_next; wt_next += (size_t)K * N;
    conv_wt_kernel<<<(K * N) / 256, 256, 0, stream>>>(Ws, dst, K, N);
    return (const _Float16*)dst;
  };
  struct LayerW { const _Float16 *Wqk,*Wv,*Wo,*W1,*W2; };
  auto prep_layer = [&](const LayerP& p, int ffn){
    LayerW lw;
    lw.Wqk = prep(p.Wqk, DMODEL, DMODEL);
    lw.Wv  = prep(p.Wv,  DMODEL, DMODEL);
    lw.Wo  = prep(p.Wo,  DMODEL, DMODEL);
    lw.W1  = ffn ? prep(p.W1, DMODEL, FFDIM) : nullptr;
    lw.W2  = ffn ? prep(p.W2, FFDIM, DMODEL) : nullptr;
    return lw;
  };
  LayerW enc1w[2] = { prep_layer(enc1[0], 1), prep_layer(enc1[1], 1) };
  LayerW enc2w[4] = { prep_layer(enc2[0], 0), prep_layer(enc2[1], 1),
                      prep_layer(enc2[2], 0), prep_layer(enc2[3], 1) };
  const _Float16* hW1t = prep(hW1, 2*DMODEL, DMODEL);
  const _Float16* hW2t = prep(hW2, DMODEL, 256);

  auto gemm = [&](const _Float16* A, const _Float16* Wt, const float* bias, void* O,
                  int M, int N, int K, int out16, int act){
    dim3 g(N / 64, M / 128);
    if (out16 && act) gemm_wmma_k<1,1><<<g, 256, 0, stream>>>(A, Wt, bias, O, M, N, K);
    else if (out16)   gemm_wmma_k<1,0><<<g, 256, 0, stream>>>(A, Wt, bias, O, M, N, K);
    else              gemm_wmma_k<0,0><<<g, 256, 0, stream>>>(A, Wt, bias, O, M, N, K);
  };

  auto run_block = [&](const LayerW& lw, const LayerP& p,
                       float* sb, _Float16* sbh, const _Float16* vsrc16,
                       int excl, int zp, int apply_pos){
    gemm(sbh,    lw.Wqk, p.bqk, QH, NTOK, DMODEL, DMODEL, 1, 0);   // q == k
    gemm(vsrc16, lw.Wv,  p.bv,  VH, NTOK, DMODEL, DMODEL, 1, 0);
    attn_kernel<<<dim3(SEQL/64, NHEAD, BATCH), 256, ATTN_LDS, stream>>>(
        QH, VH, AOh, p.gamma, excl, zp);
    gemm(AOh, lw.Wo, p.bo, A1, NTOK, DMODEL, DMODEL, 0, 0);
    ln_kernel<<<NTOK/8, 256, 0, stream>>>(sb, sbh, A1, p.ln1s, p.ln1b);
    if (apply_pos){
      gemm(sbh,  lw.W1, p.b1, F16a, NTOK, FFDIM, DMODEL, 1, 1);    // relu f16 hidden
      gemm(F16a, lw.W2, p.b2, A1,   NTOK, DMODEL, FFDIM, 0, 0);
      ln_kernel<<<NTOK/8, 256, 0, stream>>>(sb, sbh, A1, p.ln2s, p.ln2b);
    }
  };

  // ---- embeddings ----
  embed_kernel<<<NTOK, 256, 0, stream>>>(fp(0), fp(1), fp(2), fp(3), fp(4),
                                         cseq, qseq, rseq, Y, Yh, QE, QEh);

  // ---- knowledge encoder (enc1): self, incl mask, FFN ----
  run_block(enc1w[0], enc1[0], Y, Yh, Yh, 0, 0, 1);
  run_block(enc1w[1], enc1[1], Y, Yh, Yh, 0, 0, 1);

  // ---- question encoder / retriever (enc2) ----
  (void)hipMemcpyAsync(X,  QE,  SZF, hipMemcpyDeviceToDevice, stream);
  (void)hipMemcpyAsync(Xh, QEh, SZH, hipMemcpyDeviceToDevice, stream);
  run_block(enc2w[0], enc2[0], X, Xh, Xh, 0, 0, 0);   // self, incl, no FFN
  run_block(enc2w[1], enc2[1], X, Xh, Yh, 1, 1, 1);   // cross (v=y), excl, zero_pad
  run_block(enc2w[2], enc2[2], X, Xh, Xh, 0, 0, 0);
  run_block(enc2w[3], enc2[3], X, Xh, Yh, 1, 1, 1);

  // ---- head ----
  concat_kernel<<<NTOK*DMODEL/256, 256, 0, stream>>>(X, QE, F16a);   // hcat f16
  gemm(F16a, hW1t, hb1, QH, NTOK, 512, 1024, 1, 1);                  // h1 relu f16
  gemm(QH,   hW2t, hb2, VH, NTOK, 256, 512,  1, 1);                  // h2 relu f16
  head_final_kernel<<<NTOK/8, 256, 0, stream>>>(VH, hW3, hb3, (float*)d_out);
}
